// ContractToLaw_83116207112350
// MI455X (gfx1250) — matrix operand order repaired and checked by reading, not compiled, not run on hardware
//
#include <hip/hip_runtime.h>
#include <hip/hip_bf16.h>

// Problem constants (from reference)
#define NC   32          // n_contract
#define NR   256         // n_region (R)
#define NLAW 32          // n_law
#define NL   256         // max_len (L)
#define DIM  512         // EMBED_DIM
#define LDT  (NL + 8)    // attnT padded row stride (16B-aligned rows, low bank conflict)
#define SMOOTH      4.0f
#define LAMBDA_LSE  6.0f
#define LEAKY       0.1f
#define EPSV        1e-8f

typedef __attribute__((ext_vector_type(16))) __bf16 v16bf;
typedef __attribute__((ext_vector_type(8)))  __bf16 v8bf;
typedef __attribute__((ext_vector_type(2)))  __bf16 v2bf;
typedef __attribute__((ext_vector_type(8)))  float  v8f;

__device__ __forceinline__ __bf16 to_bf16(float f) {
    unsigned u = __builtin_bit_cast(unsigned, f);
    u += 0x7FFFu + ((u >> 16) & 1u);              // round-to-nearest-even
    unsigned short h = (unsigned short)(u >> 16);
    return __builtin_bit_cast(__bf16, h);
}
// Packed pair conversion: v_cvt_pk_bf16_f32 when available (1 instr / 2 values)
__device__ __forceinline__ v2bf cvt2_bf16(float a, float b) {
#if __has_builtin(__builtin_amdgcn_cvt_pk_bf16_f32)
    return __builtin_amdgcn_cvt_pk_bf16_f32(a, b);
#else
    v2bf r; r[0] = to_bf16(a); r[1] = to_bf16(b); return r;
#endif
}
__device__ __forceinline__ float from_bf16(__bf16 h) {
    unsigned short u = __builtin_bit_cast(unsigned short, h);
    return __builtin_bit_cast(float, ((unsigned)u) << 16);
}

__device__ __forceinline__ float wred_add(float v) {
    #pragma unroll
    for (int off = 16; off > 0; off >>= 1) v += __shfl_xor(v, off, 32);
    return v;
}
__device__ __forceinline__ float wred_max(float v) {
    #pragma unroll
    for (int off = 16; off > 0; off >>= 1) v = fmaxf(v, __shfl_xor(v, off, 32));
    return v;
}

// ---- WMMA bf16 fragment loaders (wave32 layouts per CDNA5 ISA 7.12.2) ----
__device__ __forceinline__ v16bf load_a_frag(const __bf16* base, int ld, int m16, int half) {
    const __bf16* p = base + m16 * ld + half * 8;
    v8bf lo = *(const v8bf*)p;
    v8bf hi = *(const v8bf*)(p + 16);
    return __builtin_shufflevector(lo, hi, 0,1,2,3,4,5,6,7,8,9,10,11,12,13,14,15);
}
// B(k,n) = src[n*ld + k]: each lane reads k=8*half..+7 and 16+8*half..+7 (2x b128)
__device__ __forceinline__ v16bf load_b_frag(const __bf16* base, int ld, int n16, int half) {
    const __bf16* p = base + n16 * ld + half * 8;
    v8bf lo = *(const v8bf*)p;
    v8bf hi = *(const v8bf*)(p + 16);
    return __builtin_shufflevector(lo, hi, 0,1,2,3,4,5,6,7,8,9,10,11,12,13,14,15);
}

#define WMMA_BF16(A, B, C) \
    __builtin_amdgcn_wmma_f32_16x16x32_bf16(false, (A), false, (B), (short)0, (C), false, false)

// Software-pipelined 1x4 tile-group GEMM over K with explicit ping-pong
// double buffering: buffers are always written by loads, never copied, so no
// rotation v_movs. First pair is peeled so the C=0 folds into the first WMMAs.
// Requires K >= 128 and K % 64 == 0.
template<int K>
__device__ __forceinline__ void mma_group(const __bf16* Ab, int lda,
                                          const __bf16* Bb, int ldb, int bstride,
                                          int m16, int half, v8f acc[4]) {
    v16bf a0, a1, b0[4], b1[4];
    // steps 0 and 1
    a0 = load_a_frag(Ab, lda, m16, half);
    #pragma unroll
    for (int q = 0; q < 4; ++q) b0[q] = load_b_frag(Bb + q * bstride, ldb, m16, half);
    a1 = load_a_frag(Ab + 32, lda, m16, half);
    #pragma unroll
    for (int q = 0; q < 4; ++q) b1[q] = load_b_frag(Bb + q * bstride + 32, ldb, m16, half);
    #pragma unroll
    for (int q = 0; q < 4; ++q) acc[q] = WMMA_BF16(a0, b0[q], acc[q]);  // acc=={} folds to 0
    a0 = load_a_frag(Ab + 64, lda, m16, half);
    #pragma unroll
    for (int q = 0; q < 4; ++q) b0[q] = load_b_frag(Bb + q * bstride + 64, ldb, m16, half);
    #pragma unroll
    for (int q = 0; q < 4; ++q) acc[q] = WMMA_BF16(a1, b1[q], acc[q]);
    // steady state: pairs (k/32, k/32+1), buf0 holds step k/32 on entry
    #pragma unroll 1
    for (int k = 64; k < K - 64; k += 64) {
        a1 = load_a_frag(Ab + k + 32, lda, m16, half);
        #pragma unroll
        for (int q = 0; q < 4; ++q) b1[q] = load_b_frag(Bb + q * bstride + k + 32, ldb, m16, half);
        #pragma unroll
        for (int q = 0; q < 4; ++q) acc[q] = WMMA_BF16(a0, b0[q], acc[q]);
        a0 = load_a_frag(Ab + k + 64, lda, m16, half);
        #pragma unroll
        for (int q = 0; q < 4; ++q) b0[q] = load_b_frag(Bb + q * bstride + k + 64, ldb, m16, half);
        #pragma unroll
        for (int q = 0; q < 4; ++q) acc[q] = WMMA_BF16(a1, b1[q], acc[q]);
    }
    // tail pair: steps (K-64)/32 in buf0, (K-32)/32 loaded now
    a1 = load_a_frag(Ab + K - 32, lda, m16, half);
    #pragma unroll
    for (int q = 0; q < 4; ++q) b1[q] = load_b_frag(Bb + q * bstride + K - 32, ldb, m16, half);
    #pragma unroll
    for (int q = 0; q < 4; ++q) acc[q] = WMMA_BF16(a0, b0[q], acc[q]);
    #pragma unroll
    for (int q = 0; q < 4; ++q) acc[q] = WMMA_BF16(a1, b1[q], acc[q]);
}

// =====================================================================
// Kernel 1: f32 -> bf16 conversion of contracts & laws, + contract row norms w1
// =====================================================================
__global__ __launch_bounds__(256)
void prep_kernel(const float* __restrict__ contracts, const float* __restrict__ laws,
                 __bf16* __restrict__ contractsB, __bf16* __restrict__ lawsB,
                 float* __restrict__ w1) {
    const int row = blockIdx.x;
    const bool isCon = row < NC * NR;
    const float* src = isCon ? (contracts + (size_t)row * DIM)
                             : (laws + (size_t)(row - NC * NR) * DIM);
    __bf16* dst = isCon ? (contractsB + (size_t)row * DIM)
                        : (lawsB + (size_t)(row - NC * NR) * DIM);
    const int i = threadIdx.x * 2;                 // DIM/2 == blockDim
    float f0 = src[i], f1 = src[i + 1];
    *(v2bf*)(dst + i) = cvt2_bf16(f0, f1);
    if (isCon) {
        float sq = wred_add(f0 * f0 + f1 * f1);
        __shared__ float red[8];
        const int wave = threadIdx.x >> 5, lane = threadIdx.x & 31;
        if (lane == 0) red[wave] = sq;
        __syncthreads();
        if (threadIdx.x == 0) {
            float t = 0.f;
            #pragma unroll
            for (int k = 0; k < 8; ++k) t += red[k];
            w1[row] = sqrtf(t);
        }
    }
}

// =====================================================================
// Kernel 2: per-law Gram matrix  G[n] = laws[n] @ laws[n]^T  (bf16 out)
// =====================================================================
__global__ __launch_bounds__(256, 2)
void gram_kernel(const __bf16* __restrict__ lawsB, __bf16* __restrict__ gramB) {
    const int n = blockIdx.x;
    const int wave = threadIdx.x >> 5, lane = threadIdx.x & 31;
    const int m16 = lane & 15, half = lane >> 4;
    const __bf16* lawN = lawsB + (size_t)n * NL * DIM;
    __bf16* G = gramB + (size_t)n * NL * NL;

    for (int g = wave; g < 64; g += 8) {          // 16 row-blocks x 4 col-groups
        const int li  = (g >> 2) * 16;
        const int ri0 = (g & 3) * 64;
        v8f acc[4] = {};
        mma_group<DIM>(lawN + li * DIM, DIM, lawN + ri0 * DIM, DIM, 16 * DIM,
                       m16, half, acc);
        #pragma unroll
        for (int q = 0; q < 4; ++q) {
            __bf16* dst = G + (size_t)(li + half * 8) * NL + ri0 + q * 16 + m16;
            #pragma unroll
            for (int j = 0; j < 8; j += 2) {
                v2bf p = cvt2_bf16(acc[q][j], acc[q][j + 1]);
                dst[(size_t)j * NL]       = p[0];
                dst[(size_t)(j + 1) * NL] = p[1];
            }
        }
    }
}

// =====================================================================
// Kernel 3: fused main kernel. One block per (n,c). 8 waves, 262KB dynamic LDS
// (LDS-limited to 1 block/WGP = 2 waves/SIMD, so grant the full VGPR budget).
// LDS: Sb bf16[NL][NR] raw scores (128KB); attnT bf16[NR][LDT] (132KB); w12/w2sq (2KB)
// =====================================================================
__global__ __launch_bounds__(256, 2)
void fused_kernel(const __bf16* __restrict__ lawsB, const __bf16* __restrict__ contractsB,
                  const __bf16* __restrict__ gramB, const float* __restrict__ w1,
                  const int* __restrict__ law_lens, float* __restrict__ out) {
    extern __shared__ char smem_raw[];
    __bf16* Sb    = (__bf16*)smem_raw;                                   // [NL][NR]
    __bf16* attnT = (__bf16*)(smem_raw + NL * NR * 2);                   // [NR][LDT]
    float*  w12   = (float*)(smem_raw + NL * NR * 2 + NR * LDT * 2);     // [NR]
    float*  w2sq  = w12 + NR;                                            // [NR]

    const int n = blockIdx.x & (NLAW - 1);
    const int c = blockIdx.x >> 5;
    const int tid = threadIdx.x;
    const int wave = tid >> 5, lane = tid & 31;
    const int m16 = lane & 15, half = lane >> 4;

    const __bf16* lawN = lawsB + (size_t)n * NL * DIM;
    const __bf16* conC = contractsB + (size_t)c * NR * DIM;
    const __bf16* G    = gramB + (size_t)n * NL * NL;

    for (int i = tid; i < 2 * NR; i += 256) w12[i] = 0.f;   // zero w12 + w2sq

    // ---------------- Phase A: raw scores -> Sb (bf16) ----------------
    for (int g = wave; g < 64; g += 8) {
        const int li  = (g >> 2) * 16;
        const int ri0 = (g & 3) * 64;
        v8f acc[4] = {};
        mma_group<DIM>(lawN + li * DIM, DIM, conC + ri0 * DIM, DIM, 16 * DIM,
                       m16, half, acc);
        #pragma unroll
        for (int q = 0; q < 4; ++q) {
            __bf16* dst = Sb + (li + half * 8) * NR + ri0 + q * 16 + m16;
            #pragma unroll
            for (int j = 0; j < 8; j += 2) {
                v2bf p = cvt2_bf16(acc[q][j], acc[q][j + 1]);
                dst[j * NR]       = p[0];
                dst[(j + 1) * NR] = p[1];
            }
        }
    }
    __syncthreads();

    // ---------------- Phase B: activation + softmax rows ----------------
    const float inv_sqrt_d = 0.044194173824159216f;  // 1/sqrt(512)
    const int len = law_lens[n];
    float w12p[8];
    #pragma unroll
    for (int j = 0; j < 8; ++j) w12p[j] = 0.f;

    for (int l = wave; l < NL; l += 8) {        // one row per wave
        const __bf16* row = Sb + l * NR;
        float sraw[8], v[8];
        float sq = 0.f;
        #pragma unroll
        for (int j = 0; j < 8; ++j) {           // r = j*32 + lane
            float s = from_bf16(row[j * 32 + lane]);
            sraw[j] = s;
            float x = s * inv_sqrt_d;
            x = (x >= 0.f) ? x : LEAKY * x;     // leaky relu
            v[j] = x;
            sq += x * x;
        }
        const float inv = 1.f / (sqrtf(wred_add(sq)) + EPSV);   // row L2 norm
        float mx = -1e30f;
        #pragma unroll
        for (int j = 0; j < 8; ++j) { v[j] *= inv; mx = fmaxf(mx, v[j]); }
        mx = wred_max(mx);
        float se = 0.f;
        #pragma unroll
        for (int j = 0; j < 8; ++j) { v[j] = __expf(v[j] - mx); se += v[j]; }
        se = wred_add(se);
        const float scale = (l < len) ? (SMOOTH / se) : 0.f;    // softmax*4, ragged mask
        #pragma unroll
        for (int j = 0; j < 8; j += 2) {
            float a0 = v[j] * scale, a1 = v[j + 1] * scale;
            w12p[j]     += a0 * sraw[j];
            w12p[j + 1] += a1 * sraw[j + 1];
            v2bf p = cvt2_bf16(a0, a1);
            attnT[((j    ) * 32 + lane) * LDT + l] = p[0];   // transposed bf16 store
            attnT[((j + 1) * 32 + lane) * LDT + l] = p[1];
        }
    }
    #pragma unroll
    for (int j = 0; j < 8; ++j) atomicAdd(&w12[j * 32 + lane], w12p[j]);
    __syncthreads();

    // ------- Phase C: w2sq[r] = attn[:,r]^T G attn[:,r]  via T = G @ attn -------
    for (int g = wave; g < 64; g += 8) {
        const int li  = (g >> 2) * 16;
        const int ri0 = (g & 3) * 64;
        v8f acc[4] = {};
        mma_group<NL>(G + li * NL, NL, attnT + ri0 * LDT, LDT, 16 * LDT,
                      m16, half, acc);
        // diag combine: acc[q][j] is T(li+half*8+j, r), r = ri0+q*16+m16;
        // attn(l, r) = attnT[r][l]: 8 consecutive bf16 -> one b128 DS load.
        #pragma unroll
        for (int q = 0; q < 4; ++q) {
            const int r = ri0 + q * 16 + m16;
            v8bf av = *(const v8bf*)(attnT + r * LDT + li + 8 * half);
            float part = 0.f;
            #pragma unroll
            for (int j = 0; j < 8; ++j) part += acc[q][j] * from_bf16(av[j]);
            atomicAdd(&w2sq[r], part);
        }
    }
    __syncthreads();

    // ---------------- Phase D: cosine + LSE over regions ----------------
    if (wave == 0) {
        float vals[8], mx = -1e30f;
        #pragma unroll
        for (int j = 0; j < 8; ++j) {
            const int r = j * 32 + lane;
            float a1 = w1[c * NR + r];
            float a2 = sqrtf(fmaxf(w2sq[r], 0.f));
            float sim = w12[r] / fmaxf(a1 * a2, EPSV);
            vals[j] = sim * LAMBDA_LSE;
            mx = fmaxf(mx, vals[j]);
        }
        mx = wred_max(mx);
        float se = 0.f;
        #pragma unroll
        for (int j = 0; j < 8; ++j) se += __expf(vals[j] - mx);
        se = wred_add(se);
        if (lane == 0) out[c * NLAW + n] = (mx + logf(se)) / LAMBDA_LSE;
    }
}

// =====================================================================
// Launcher. Workspace: lawsB 8MB | contractsB 8MB | gramB 4MB | w1 32KB (~21MB)
// =====================================================================
extern "C" void kernel_launch(void* const* d_in, const int* in_sizes, int n_in,
                              void* d_out, int out_size, void* d_ws, size_t ws_size,
                              hipStream_t stream) {
    const float* contracts = (const float*)d_in[0];
    const float* laws      = (const float*)d_in[1];
    const int*   law_lens  = (const int*)d_in[2];
    float* out = (float*)d_out;

    char* ws = (char*)d_ws;
    __bf16* lawsB      = (__bf16*)(ws);
    __bf16* contractsB = (__bf16*)(ws + (size_t)NLAW * NL * DIM * 2);
    __bf16* gramB      = (__bf16*)(ws + (size_t)(NLAW * NL + NC * NR) * DIM * 2);
    float*  w1         = (float*)(ws + (size_t)(NLAW * NL + NC * NR) * DIM * 2
                                      + (size_t)NLAW * NL * NL * 2);

    prep_kernel<<<NC * NR + NLAW * NL, 256, 0, stream>>>(contracts, laws,
                                                         contractsB, lawsB, w1);
    gram_kernel<<<NLAW, 256, 0, stream>>>(lawsB, gramB);

    const size_t smem = (size_t)NL * NR * 2      // Sb
                      + (size_t)NR * LDT * 2     // attnT
                      + 2 * NR * 4;              // w12 + w2sq   = 262 KB < 320 KB
    fused_kernel<<<NC * NLAW, 256, smem, stream>>>(lawsB, contractsB, gramB,
                                                   w1, law_lens, out);
}
